// DirectVoxGO_26886495273778
// MI455X (gfx1250) — matrix operand order repaired and checked by reading, not compiled, not run on hardware
//
#include <hip/hip_runtime.h>
#include <math.h>

typedef _Float16 half_t;
typedef __attribute__((ext_vector_type(16))) _Float16 v16h;
typedef __attribute__((ext_vector_type(8)))  _Float16 v8h;
typedef __attribute__((ext_vector_type(8)))  float    v8f;

// ---------------- problem constants (fixed by the reference) ----------------
constexpr int   N_RAYS   = 8192;
constexpr int   M_PTS    = 1048576;
constexpr int   GS       = 160;
constexpr long  GS3      = (long)GS * GS * GS;
constexpr int   K0_DIM   = 12;
constexpr int   WIDTH    = 128;
constexpr int   DIM0     = 39;          // 3 + 3*4*2 + 12
constexpr int   K0PAD    = 64;          // layer-0 K padded for 2x 16x16x32 WMMA
constexpr float ACT_SHIFT = -4.59511985013459f;   // log(1/(1-0.01)-1)
constexpr float XYZ_MIN = -1.0f, XYZ_MAX = 1.0f;

// f16 weight blob layout (contiguous in workspace, copied to LDS as one block)
constexpr int W0T_HALFS = WIDTH * K0PAD;     // [n=128][k=64]
constexpr int W1T_HALFS = WIDTH * WIDTH;     // [n=128][k=128]
constexpr int W2T_HALFS = 16 * WIDTH;        // [n=16 (3 real)][k=128]
constexpr int WALL_HALFS = W0T_HALFS + W1T_HALFS + W2T_HALFS;   // 26624
constexpr int BALL_FLOATS = 128 + 128 + 16;  // f32 biases, b2 zero-padded to 16

// MLP kernel LDS layout (halfs), 8 waves/block, 16 points/wave
constexpr int SX_HALFS  = 8 * 16 * K0PAD;    // per-wave x tiles
constexpr int SH_HALFS  = 8 * 16 * WIDTH;    // per-wave h tiles
constexpr int SMEM_HALFS = WALL_HALFS + 2 * BALL_FLOATS + SX_HALFS + 2 * SH_HALFS;

// ---------------------------------------------------------------------------
// Kernel 0: weights -> f16 transposed [n][k] (zero-padded) + f32 bias blob
// ---------------------------------------------------------------------------
__global__ void prep_weights_kernel(const float* __restrict__ w0,
                                    const float* __restrict__ w1,
                                    const float* __restrict__ w2,
                                    const float* __restrict__ b0,
                                    const float* __restrict__ b1,
                                    const float* __restrict__ b2,
                                    half_t* __restrict__ wAll,
                                    float* __restrict__ bAll) {
    int i = blockIdx.x * blockDim.x + threadIdx.x;
    if (i < W0T_HALFS) {
        int n = i / K0PAD, k = i % K0PAD;
        wAll[i] = (half_t)((k < DIM0) ? w0[k * WIDTH + n] : 0.0f);
    } else if (i < W0T_HALFS + W1T_HALFS) {
        int j = i - W0T_HALFS;
        int n = j / WIDTH, k = j % WIDTH;
        wAll[i] = (half_t)w1[k * WIDTH + n];
    } else if (i < WALL_HALFS) {
        int j = i - W0T_HALFS - W1T_HALFS;
        int n = j / WIDTH, k = j % WIDTH;
        wAll[i] = (half_t)((n < 3) ? w2[k * 3 + n] : 0.0f);
    } else if (i < WALL_HALFS + BALL_FLOATS) {
        int j = i - WALL_HALFS;
        float v;
        if (j < 128)      v = b0[j];
        else if (j < 256) v = b1[j - 128];
        else              v = (j - 256 < 3) ? b2[j - 256] : 0.0f;
        bAll[j] = v;
    }
}

// ---------------------------------------------------------------------------
// Kernel 1: per-ray viewdir positional embedding -> 32 f16 (27 real + pad)
// order matches jnp: [vd(3), sin(vd[c]*2^f) c-major (12), cos (12)]
// ---------------------------------------------------------------------------
__global__ void prep_rays_kernel(const float* __restrict__ vd,
                                 half_t* __restrict__ vembH) {
    int r = blockIdx.x * blockDim.x + threadIdx.x;
    if (r >= N_RAYS) return;
    float v0 = vd[r * 3 + 0], v1 = vd[r * 3 + 1], v2 = vd[r * 3 + 2];
    half_t e[32];
#pragma unroll
    for (int i = 0; i < 32; ++i) e[i] = (half_t)0.0f;
    e[0] = (half_t)v0; e[1] = (half_t)v1; e[2] = (half_t)v2;
    float vv[3] = {v0, v1, v2};
#pragma unroll
    for (int c = 0; c < 3; ++c) {
#pragma unroll
        for (int f = 0; f < 4; ++f) {
            float ang = vv[c] * (float)(1 << f);
            e[3 + c * 4 + f]  = (half_t)sinf(ang);
            e[15 + c * 4 + f] = (half_t)cosf(ang);
        }
    }
    v8h* dst = (v8h*)(vembH + (size_t)r * 32);
#pragma unroll
    for (int q = 0; q < 4; ++q) dst[q] = *(const v8h*)(e + q * 8);
}

// ---------------------------------------------------------------------------
// Kernel 2: per-point trilerp of density (-> log(1-alpha)) and k0 features
// ---------------------------------------------------------------------------
__global__ void sample_kernel(const float* __restrict__ pts,
                              const float* __restrict__ dgrid,
                              const float* __restrict__ k0g,
                              float* __restrict__ logm,
                              half_t* __restrict__ featH) {
    int p = blockIdx.x * blockDim.x + threadIdx.x;
    if (p >= M_PTS) return;

    int   i0[3]; float fr[3];
#pragma unroll
    for (int d = 0; d < 3; ++d) {
        float x = pts[(size_t)p * 3 + d];
        float t = (x - XYZ_MIN) / (XYZ_MAX - XYZ_MIN) * (float)(GS - 1);
        t = fminf(fmaxf(t, 0.0f), (float)(GS - 1));
        int i = (int)floorf(t);
        if (i > GS - 2) i = GS - 2;
        i0[d] = i;
        fr[d] = t - (float)i;
    }
    float fx = fr[0], fy = fr[1], fz = fr[2];
    float wc[8] = {(1-fx)*(1-fy)*(1-fz), (1-fx)*(1-fy)*fz,
                   (1-fx)*fy*(1-fz),     (1-fx)*fy*fz,
                   fx*(1-fy)*(1-fz),     fx*(1-fy)*fz,
                   fx*fy*(1-fz),         fx*fy*fz};
    int xs[2] = {i0[0], i0[0] + 1}, ys[2] = {i0[1], i0[1] + 1}, zs[2] = {i0[2], i0[2] + 1};

    // warm the four grid lines we are about to gather (global_prefetch_b8)
#pragma unroll
    for (int cx = 0; cx < 2; ++cx)
#pragma unroll
        for (int cy = 0; cy < 2; ++cy) {
            size_t b = ((size_t)xs[cx] * GS + ys[cy]) * GS + zs[0];
            __builtin_prefetch(dgrid + b, 0, 3);
            __builtin_prefetch(k0g + b, 0, 3);
        }

    float raw = 0.0f;
    float acc[K0_DIM];
#pragma unroll
    for (int c = 0; c < K0_DIM; ++c) acc[c] = 0.0f;

#pragma unroll
    for (int cx = 0; cx < 2; ++cx)
#pragma unroll
        for (int cy = 0; cy < 2; ++cy)
#pragma unroll
            for (int cz = 0; cz < 2; ++cz) {
                int corner = (cx << 2) | (cy << 1) | cz;
                size_t base = ((size_t)xs[cx] * GS + ys[cy]) * GS + zs[cz];
                float w = wc[corner];
                raw += w * dgrid[base];
                const float* kp = k0g + base;
#pragma unroll
                for (int c = 0; c < K0_DIM; ++c)
                    acc[c] += w * kp[(size_t)c * GS3];
            }

    // alpha = 1 - exp(-softplus(raw+shift))  =>  log(1-alpha) = -softplus(.)
    float spin = raw + ACT_SHIFT;
    float sp = (spin > 20.0f) ? spin : log1pf(expf(spin));
    logm[p] = -sp;

    half_t fh[16];
#pragma unroll
    for (int c = 0; c < K0_DIM; ++c) fh[c] = (half_t)acc[c];
#pragma unroll
    for (int c = K0_DIM; c < 16; ++c) fh[c] = (half_t)0.0f;
    v8h* dst = (v8h*)(featH + (size_t)p * 16);
    dst[0] = *(const v8h*)(fh);
    dst[1] = *(const v8h*)(fh + 8);
}

// ---------------------------------------------------------------------------
// Kernel 3: per-ray serial transmittance scan (ray_id is sorted)
// ---------------------------------------------------------------------------
__device__ __forceinline__ int lower_bound_rid(const int* __restrict__ rid, int key) {
    int lo = 0, hi = M_PTS;
    while (lo < hi) {
        int mid = (lo + hi) >> 1;
        if (rid[mid] < key) lo = mid + 1; else hi = mid;
    }
    return lo;
}

__global__ void scan_kernel(const int* __restrict__ rid,
                            const float* __restrict__ logm,
                            float* __restrict__ wts,
                            float* __restrict__ ainv) {
    int r = blockIdx.x * blockDim.x + threadIdx.x;
    if (r >= N_RAYS) return;
    int s = lower_bound_rid(rid, r);
    int e = lower_bound_rid(rid, r + 1);
    float cum = 0.0f;   // exclusive cumsum of log(1-alpha) inside the segment
    for (int i = s; i < e; ++i) {
        float l = logm[i];
        wts[i] = (1.0f - expf(l)) * expf(cum);   // alpha * T
        cum += l;
    }
    ainv[r] = expf(cum);                          // alphainv_last
}

// ---------------------------------------------------------------------------
// Kernel 4: WMMA MLP.  256 threads = 8 waves, 16 points per wave.
// A fragments hoisted per layer; bias pre-loaded into the C accumulator.
// ---------------------------------------------------------------------------
__device__ __forceinline__ v16h load_a_frag(const half_t* __restrict__ tile,
                                            int m, int kc, int lane, int stride) {
    // f16 A 16x32 layout: lanes 0-15: K0..7 | K16..23 ; lanes 16-31: K8..15 | K24..31
    int off = (lane >= 16) ? 8 : 0;
    const half_t* row = tile + m * stride + kc * 32 + off;
    v8h lo = *(const v8h*)(row);
    v8h hi = *(const v8h*)(row + 16);
    v16h a;
#pragma unroll
    for (int i = 0; i < 8; ++i) { a[i] = lo[i]; a[i + 8] = hi[i]; }
    return a;
}

__device__ __forceinline__ v16h load_b_frag(const half_t* __restrict__ wt,
                                            int n, int kc, int lane, int Kdim) {
    // f16 B 32x16 layout: lane holds one N column; lanes 0-15: K0..7 | K16..23 ; lanes 16-31: K8..15 | K24..31
    int off = (lane >= 16) ? 8 : 0;
    const half_t* col = wt + n * Kdim + kc * 32 + off;
    v8h lo = *(const v8h*)(col);
    v8h hi = *(const v8h*)(col + 16);
    v16h b;
#pragma unroll
    for (int i = 0; i < 8; ++i) { b[i] = lo[i]; b[i + 8] = hi[i]; }
    return b;
}

__launch_bounds__(256)
__global__ void mlp_kernel(const half_t* __restrict__ featH,
                           const half_t* __restrict__ vembH,
                           const half_t* __restrict__ wAll,
                           const float* __restrict__ bAll,
                           const float* __restrict__ wts,
                           const int* __restrict__ rid,
                           float* __restrict__ wrgb) {
    extern __shared__ __align__(16) char smem_raw[];
    half_t* lds = (half_t*)smem_raw;
    half_t* sw0 = lds;                      // [128][64]
    half_t* sw1 = sw0 + W0T_HALFS;          // [128][128]
    half_t* sw2 = sw1 + W1T_HALFS;          // [16][128]
    float*  sb  = (float*)(sw2 + W2T_HALFS);        // 272 f32 biases
    half_t* sx  = (half_t*)(sb + BALL_FLOATS);      // 8 x [16][64]
    half_t* sh1 = sx + SX_HALFS;            // 8 x [16][128]
    half_t* sh2 = sh1 + SH_HALFS;           // 8 x [16][128]

    const int tid  = threadIdx.x;
    const int lane = tid & 31;
    const int wave = tid >> 5;
    const int tilebase = blockIdx.x * 128 + wave * 16;

    // cooperative copy of weights + biases into LDS (dword-coalesced)
    {
        const unsigned* src = (const unsigned*)wAll;
        unsigned* dst = (unsigned*)lds;
        for (int i = tid; i < WALL_HALFS / 2; i += 256) dst[i] = src[i];
        for (int i = tid; i < BALL_FLOATS; i += 256) sb[i] = bAll[i];
    }

    // build this wave's x tile: [feat(12) | vemb[rid](27) | 0-pad] per point
    half_t* xt = sx + wave * 16 * K0PAD;
    {
        int mr = lane & 15, part = lane >> 4;       // lane covers 32 columns
        int p = tilebase + mr;
        int r = rid[p];
        const half_t* fp = featH + (size_t)p * 16;
        const half_t* vp = vembH + (size_t)r * 32;
        half_t tmp[32];
#pragma unroll
        for (int k = 0; k < 32; ++k) {
            int c = part * 32 + k;
            tmp[k] = (c < K0_DIM) ? fp[c]
                   : (c < DIM0)   ? vp[c - K0_DIM]
                                  : (half_t)0.0f;
        }
        v8h* dst = (v8h*)(xt + mr * K0PAD + part * 32);
#pragma unroll
        for (int q = 0; q < 4; ++q) dst[q] = *(const v8h*)(tmp + q * 8);
    }
    __syncthreads();

    half_t* h1 = sh1 + wave * 16 * WIDTH;
    half_t* h2 = sh2 + wave * 16 * WIDTH;
    const float* sb0 = sb;
    const float* sb1 = sb + 128;
    const float* sb2 = sb + 256;
    const int m = lane & 15;
    const int n = lane & 15;

    // ---- layer 0: 16x64 @ 64x128, relu ----
    {
        v16h a0 = load_a_frag(xt, m, 0, lane, K0PAD);
        v16h a1 = load_a_frag(xt, m, 1, lane, K0PAD);
#pragma unroll
        for (int j = 0; j < 8; ++j) {
            float bias = sb0[j * 16 + n];
            v8f acc;
#pragma unroll
            for (int r8 = 0; r8 < 8; ++r8) acc[r8] = bias;   // bias as C init
            acc = __builtin_amdgcn_wmma_f32_16x16x32_f16(false, a0, false,
                    load_b_frag(sw0, j * 16 + n, 0, lane, K0PAD),
                    (short)0, acc, false, false);
            acc = __builtin_amdgcn_wmma_f32_16x16x32_f16(false, a1, false,
                    load_b_frag(sw0, j * 16 + n, 1, lane, K0PAD),
                    (short)0, acc, false, false);
#pragma unroll
            for (int r8 = 0; r8 < 8; ++r8) {
                int mrow = (lane < 16) ? r8 : (r8 + 8);
                float v = acc[r8];
                v = v > 0.0f ? v : 0.0f;
                h1[mrow * WIDTH + j * 16 + n] = (half_t)v;
            }
        }
    }
    __syncthreads();

    // ---- layer 1: 16x128 @ 128x128, relu ----
    {
        v16h a[4];
#pragma unroll
        for (int kc = 0; kc < 4; ++kc) a[kc] = load_a_frag(h1, m, kc, lane, WIDTH);
#pragma unroll
        for (int j = 0; j < 8; ++j) {
            float bias = sb1[j * 16 + n];
            v8f acc;
#pragma unroll
            for (int r8 = 0; r8 < 8; ++r8) acc[r8] = bias;
#pragma unroll
            for (int kc = 0; kc < 4; ++kc) {
                acc = __builtin_amdgcn_wmma_f32_16x16x32_f16(false, a[kc], false,
                        load_b_frag(sw1, j * 16 + n, kc, lane, WIDTH),
                        (short)0, acc, false, false);
            }
#pragma unroll
            for (int r8 = 0; r8 < 8; ++r8) {
                int mrow = (lane < 16) ? r8 : (r8 + 8);
                float v = acc[r8];
                v = v > 0.0f ? v : 0.0f;
                h2[mrow * WIDTH + j * 16 + n] = (half_t)v;
            }
        }
    }
    __syncthreads();

    // ---- layer 2: 16x128 @ 128x16 (3 real cols), sigmoid, weight, store ----
    {
        float bias = sb2[n];
        v8f acc;
#pragma unroll
        for (int r8 = 0; r8 < 8; ++r8) acc[r8] = bias;
#pragma unroll
        for (int kc = 0; kc < 4; ++kc) {
            v16h a = load_a_frag(h2, m, kc, lane, WIDTH);
            acc = __builtin_amdgcn_wmma_f32_16x16x32_f16(false, a, false,
                    load_b_frag(sw2, n, kc, lane, WIDTH),
                    (short)0, acc, false, false);
        }
        if (n < 3) {
#pragma unroll
            for (int r8 = 0; r8 < 8; ++r8) {
                int mrow = (lane < 16) ? r8 : (r8 + 8);
                int p = tilebase + mrow;
                float rgb = 1.0f / (1.0f + expf(-acc[r8]));
                wrgb[(size_t)p * 3 + n] = wts[p] * rgb;
            }
        }
    }
}

// ---------------------------------------------------------------------------
// Kernel 5: per-ray reduction of w*rgb, plus alphainv_last  (deterministic)
// ---------------------------------------------------------------------------
__global__ void reduce_kernel(const int* __restrict__ rid,
                              const float* __restrict__ wrgb,
                              const float* __restrict__ ainv,
                              float* __restrict__ out) {
    int r = blockIdx.x * blockDim.x + threadIdx.x;
    if (r >= N_RAYS) return;
    int s = lower_bound_rid(rid, r);
    int e = lower_bound_rid(rid, r + 1);
    float s0 = 0.0f, s1 = 0.0f, s2 = 0.0f;
    for (int i = s; i < e; ++i) {
        const float* w = wrgb + (size_t)i * 3;
        s0 += w[0]; s1 += w[1]; s2 += w[2];
    }
    float a = ainv[r];
    out[r * 3 + 0] = s0 + a;
    out[r * 3 + 1] = s1 + a;
    out[r * 3 + 2] = s2 + a;
}

// ---------------------------------------------------------------------------
extern "C" void kernel_launch(void* const* d_in, const int* in_sizes, int n_in,
                              void* d_out, int out_size, void* d_ws, size_t ws_size,
                              hipStream_t stream) {
    (void)in_sizes; (void)n_in; (void)out_size; (void)ws_size;
    const float* ray_pts  = (const float*)d_in[0];
    const float* viewdirs = (const float*)d_in[1];
    const float* density  = (const float*)d_in[2];
    const float* k0       = (const float*)d_in[3];
    const float* w0       = (const float*)d_in[4];
    const float* b0       = (const float*)d_in[5];
    const float* w1       = (const float*)d_in[6];
    const float* b1       = (const float*)d_in[7];
    const float* w2       = (const float*)d_in[8];
    const float* b2       = (const float*)d_in[9];
    const int*   ray_id   = (const int*)d_in[10];
    float* out = (float*)d_out;

    // workspace carve-up (256B aligned), ~55 MB total
    char* ws = (char*)d_ws;
    size_t off = 0;
    auto take = [&](size_t bytes) {
        size_t o = off;
        off = (off + bytes + 255) & ~(size_t)255;
        return o;
    };
    half_t* vembH = (half_t*)(ws + take((size_t)N_RAYS * 32 * sizeof(half_t)));
    half_t* featH = (half_t*)(ws + take((size_t)M_PTS * 16 * sizeof(half_t)));
    float*  logm  = (float*)(ws + take((size_t)M_PTS * sizeof(float)));
    float*  wts   = (float*)(ws + take((size_t)M_PTS * sizeof(float)));
    float*  ainv  = (float*)(ws + take((size_t)N_RAYS * sizeof(float)));
    half_t* wAll  = (half_t*)(ws + take((size_t)WALL_HALFS * sizeof(half_t)));
    float*  bAll  = (float*)(ws + take((size_t)BALL_FLOATS * sizeof(float)));
    float*  wrgb  = (float*)(ws + take((size_t)M_PTS * 3 * sizeof(float)));

    prep_weights_kernel<<<(WALL_HALFS + BALL_FLOATS + 255) / 256, 256, 0, stream>>>(
        w0, w1, w2, b0, b1, b2, wAll, bAll);
    prep_rays_kernel<<<N_RAYS / 256, 256, 0, stream>>>(viewdirs, vembH);
    sample_kernel<<<M_PTS / 256, 256, 0, stream>>>(ray_pts, density, k0, logm, featH);
    scan_kernel<<<N_RAYS / 256, 256, 0, stream>>>(ray_id, logm, wts, ainv);
    mlp_kernel<<<M_PTS / 128, 256, SMEM_HALFS * sizeof(half_t), stream>>>(
        featH, vembH, wAll, bAll, wts, ray_id, wrgb);
    reduce_kernel<<<N_RAYS / 256, 256, 0, stream>>>(ray_id, wrgb, ainv, out);
}
